// Attention_90512140796786
// MI455X (gfx1250) — matrix operand order repaired and checked by reading, not compiled, not run on hardware
//
#include <hip/hip_runtime.h>
#include <hip/hip_bf16.h>

typedef __attribute__((ext_vector_type(16))) _Float16 v16h;
typedef __attribute__((ext_vector_type(8)))  _Float16 v8h;
typedef __attribute__((ext_vector_type(8)))  float    v8f;
typedef __attribute__((ext_vector_type(4)))  unsigned int v4u;
typedef __attribute__((ext_vector_type(4)))  int      v4i;
typedef __attribute__((ext_vector_type(8)))  int      v8i;

#define BATCH 4
#define SEQ   2048
#define CH    1024
#define HEADS 16
#define HDIM  64
#define BN    (BATCH*SEQ)   // 8192 rows
#define QKV_N (3*CH)        // 3072 qkv columns

// Padded LDS panel: 64 columns x CH halves; +8 halves (16B) after each 512-half chunk.
// LDS half index = col*1040 + k + 8*(k>=512)   (column stride 1040 halves = 2080B)
#define COL_STRIDE_H   (CH + 16)                  // 1040
#define PANEL_HALVES   (64 * CH)
#define PANEL_LDS_BYTES (64 * COL_STRIDE_H * 2)   // 133120

#if __has_builtin(__builtin_amdgcn_tensor_load_to_lds)
#define HAVE_TDM 1
#else
#define HAVE_TDM 0
#endif

union F16Frag { v16h v; struct { v8h lo, hi; } p; };

// A operand 16x32 f16 (row-major source, 16 consecutive rows):
// lane<16: row=lane,   K {0..15};  lane>=16: row=lane-16, K {16..31}
__device__ __forceinline__ v16h load_fragA(const _Float16* __restrict__ base, int ld, int lane) {
  const int r  = lane & 15;
  const int k8 = (lane >> 4) << 4;
  const _Float16* p = base + (size_t)r * ld + k8;
  F16Frag f;
  f.p.lo = *(const v8h*)(p);
  f.p.hi = *(const v8h*)(p + 8);
  return f.v;
}

// B operand 32x16 f16 from column-contiguous global storage
__device__ __forceinline__ v16h load_fragB(const _Float16* __restrict__ base, int ld, int lane) {
  const int n   = lane & 15;
  const int k16 = (lane >> 4) << 4;
  const _Float16* p = base + (size_t)n * ld + k16;
  F16Frag f;
  f.p.lo = *(const v8h*)(p);
  f.p.hi = *(const v8h*)(p + 8);
  return f.v;
}

// B fragment given a fully-resolved lane pointer (16 contiguous halves)
__device__ __forceinline__ v16h fragB_at(const _Float16* p) {
  F16Frag f;
  f.p.lo = *(const v8h*)(p);
  f.p.hi = *(const v8h*)(p + 8);
  return f.v;
}

__device__ __forceinline__ v8f wmma16(v16h a, v16h b, v8f c) {
  return __builtin_amdgcn_wmma_f32_16x16x32_f16(false, a, false, b, (short)0, c, false, false);
}

__device__ __forceinline__ float redmax16(float v) {
#pragma unroll
  for (int m = 1; m < 16; m <<= 1) v = fmaxf(v, __shfl_xor(v, m, 32));
  return v;
}
__device__ __forceinline__ float redsum16(float v) {
#pragma unroll
  for (int m = 1; m < 16; m <<= 1) v += __shfl_xor(v, m, 32);
  return v;
}

// Stage a 64-row x CH-half panel (rows n0.. of wT) into padded LDS via the TDM.
__device__ __forceinline__ void stage_panel(const _Float16* __restrict__ gsrc,
                                            _Float16* Bp, int tensor_rows) {
#if HAVE_TDM
  if (threadIdx.x < 32) {
    const unsigned long long ga = (unsigned long long)(uintptr_t)gsrc;
    const unsigned lds_off = (unsigned)(uintptr_t)&Bp[0];   // low 32 bits = LDS byte offset
    v4u g0;
    g0[0] = 1u;                                             // count=1, user mode
    g0[1] = lds_off;                                        // lds_addr
    g0[2] = (unsigned)(ga & 0xffffffffu);                   // global_addr[31:0]
    g0[3] = (unsigned)((ga >> 32) & 0x01ffffffu) | (2u << 30);  // ga[56:32] | type=2
    v8i g1;
    // data_size=2B(1), pad_enable, pad_interval=256dw(7), pad_amount=4dw(3)
    g1[0] = (int)((1u << 16) | (1u << 20) | (7u << 22) | (3u << 25));
    g1[1] = (int)((unsigned)(CH & 0xffff) << 16);           // tensor_dim0[15:0]=1024
    g1[2] = (int)(((unsigned)CH >> 16) | ((unsigned)(tensor_rows & 0xffff) << 16)); // dim0 hi | dim1 lo
    g1[3] = (int)(((unsigned)tensor_rows >> 16) | ((unsigned)CH << 16));            // dim1 hi | tile_dim0=1024
    g1[4] = 64;                                             // tile_dim1=64, tile_dim2=0
    g1[5] = CH;                                             // tensor_dim0_stride=1024
    g1[6] = 0;
    g1[7] = 0;
    v4i z4 = {0, 0, 0, 0};
#if __clang_major__ >= 23
    v8i z8 = {0, 0, 0, 0, 0, 0, 0, 0};
    __builtin_amdgcn_tensor_load_to_lds(g0, g1, z4, z4, z8, 0);
#else
    __builtin_amdgcn_tensor_load_to_lds(g0, g1, z4, z4, 0);
#endif
    __builtin_amdgcn_s_wait_tensorcnt(0);
  }
  __syncthreads();
#else
  // cooperative fill with the same padded layout
  for (int e = threadIdx.x * 8; e < PANEL_HALVES; e += 256 * 8) {
    v8h d = *(const v8h*)(gsrc + e);
    *(v8h*)(Bp + e + ((e >> 9) << 3)) = d;
  }
  __syncthreads();
  (void)tensor_rows;
#endif
}

// Shared inner loop: 16x64 tile accumulation, B fragments from padded LDS panel.
// All 4 B fragments are loaded into independent registers before the WMMAs so
// the ds_loads stay in flight (progressive s_wait_dscnt instead of full stalls).
__device__ __forceinline__ void gemm_tile_lds(const _Float16* __restrict__ arow,
                                              const _Float16* Bp, int lane,
                                              v8f acc[4]) {
  const int nl  = lane & 15;
  const int k16 = (lane >> 4) << 4;
  const _Float16* bb0 = Bp + (size_t)nl * COL_STRIDE_H + k16;
  const _Float16* bb1 = bb0 + 16 * COL_STRIDE_H;
  const _Float16* bb2 = bb0 + 32 * COL_STRIDE_H;
  const _Float16* bb3 = bb0 + 48 * COL_STRIDE_H;
#pragma unroll
  for (int half = 0; half < 2; ++half) {
    const int koff = half * 512;
    const int pad  = half * 8;       // constant within each half -> immediate ds offsets
#pragma unroll 4
    for (int k0 = 0; k0 < 512; k0 += 32) {
      const int o = koff + k0 + pad;
      v16h a  = load_fragA(arow + koff + k0, CH, lane);
      v16h f0 = fragB_at(bb0 + o);
      v16h f1 = fragB_at(bb1 + o);
      v16h f2 = fragB_at(bb2 + o);
      v16h f3 = fragB_at(bb3 + o);
      acc[0] = wmma16(a, f0, acc[0]);
      acc[1] = wmma16(a, f1, acc[1]);
      acc[2] = wmma16(a, f2, acc[2]);
      acc[3] = wmma16(a, f3, acc[3]);
    }
  }
}

// ---------------- prep kernels ----------------
__global__ void k_cvt_f16(const float* __restrict__ in, _Float16* __restrict__ out, int n) {
  int i = blockIdx.x * blockDim.x + threadIdx.x;
  if (i < n) out[i] = (_Float16)in[i];
}

// in [rows][cols] fp32 -> out [cols][rows] f16
__global__ void k_transpose_f16(const float* __restrict__ in, _Float16* __restrict__ out,
                                int rows, int cols) {
  int i = blockIdx.x * blockDim.x + threadIdx.x;
  if (i < rows * cols) {
    int oc   = i % rows;   // input row
    int orow = i / rows;   // input col
    out[(size_t)orow * rows + oc] = (_Float16)in[(size_t)oc * cols + orow];
  }
}

// ---------------- QKV GEMM: block stages 64-col weight panel via TDM; 8 waves share it.
// Routes q->Q [B,H,N,D] (pre-scaled), k->K [B,H,N,D], v->V^T [B,H,D,N] ----------------
__global__ void __launch_bounds__(256) k_qkv_gemm(
    const _Float16* __restrict__ xh,    // [BN][CH]
    const _Float16* __restrict__ wT,    // [3CH][CH]
    const float*    __restrict__ bias,  // [3CH]
    _Float16* __restrict__ Qd, _Float16* __restrict__ Kd, _Float16* __restrict__ Vt) {
  extern __shared__ _Float16 Bp[];
  const int NTN = QKV_N / 64;              // 48 column-panels
  const int tn  = blockIdx.x % NTN;
  const int tm  = (blockIdx.x / NTN) * 8 + (threadIdx.x >> 5);
  const int n0  = tn * 64;
  const int lane = threadIdx.x & 31;
  const int hi = lane >> 4, nl = lane & 15;

  stage_panel(wT + (size_t)n0 * CH, Bp, QKV_N);

  v8f acc[4] = {};
  gemm_tile_lds(xh + (size_t)(tm * 16) * CH, Bp, lane, acc);

  const float scale = 0.125f;  // HDIM^-0.5
#pragma unroll
  for (int j = 0; j < 4; ++j) {
    const int c  = n0 + j * 16 + nl;
    const float bv = bias[c];
    const int seg = c >> 10;             // 0=q 1=k 2=v
    const int cc = c & (CH - 1);
    const int h = cc >> 6, d = cc & 63;
#pragma unroll
    for (int jj = 0; jj < 8; ++jj) {
      const int row  = tm * 16 + jj + 8 * hi;
      const int bidx = row >> 11;
      const int nq   = row & (SEQ - 1);
      float v = acc[j][jj] + bv;
      if (seg == 0)      Qd[((size_t)(bidx * HEADS + h) * SEQ + nq) * HDIM + d] = (_Float16)(v * scale);
      else if (seg == 1) Kd[((size_t)(bidx * HEADS + h) * SEQ + nq) * HDIM + d] = (_Float16)v;
      else               Vt[((size_t)(bidx * HEADS + h) * HDIM + d) * SEQ + nq] = (_Float16)v;
    }
  }
}

// ---------------- flash attention: one wave per 16-query tile, stream 32 keys/step ----------------
__global__ void __launch_bounds__(256) k_flash(
    const _Float16* __restrict__ Qd, const _Float16* __restrict__ Kd,
    const _Float16* __restrict__ Vt, _Float16* __restrict__ attnh /* [BN][CH] */) {
  __shared__ __align__(16) _Float16 Plds[8][16 * 32];
  const int lane = threadIdx.x & 31;
  const int wid  = threadIdx.x >> 5;
  const int task = blockIdx.x * 8 + wid;  // 8192 tasks
  const int mt = task & 127;
  const int bh = task >> 7;
  const int b = bh >> 4, h = bh & 15;
  const int hi = lane >> 4, nl = lane & 15;

  const _Float16* Qb = Qd + ((size_t)bh * SEQ + (size_t)mt * 16) * HDIM;
  const _Float16* Kb = Kd + (size_t)bh * SEQ * HDIM;
  const _Float16* Vb = Vt + (size_t)bh * HDIM * SEQ;

  v16h qa0 = load_fragA(Qb + 0,  HDIM, lane);
  v16h qa1 = load_fragA(Qb + 32, HDIM, lane);

  v8f o0 = {}, o1 = {}, o2 = {}, o3 = {};
  float mrow[8], lrow[8];
#pragma unroll
  for (int jj = 0; jj < 8; ++jj) { mrow[jj] = -3.0e38f; lrow[jj] = 0.f; }

  for (int kt = 0; kt < SEQ; kt += 32) {
    v8f z = {};
    v8f s0 = wmma16(qa1, load_fragB(Kb + (size_t)kt * HDIM + 32, HDIM, lane),
             wmma16(qa0, load_fragB(Kb + (size_t)kt * HDIM + 0,  HDIM, lane), z));
    v8f s1 = wmma16(qa1, load_fragB(Kb + (size_t)(kt + 16) * HDIM + 32, HDIM, lane),
             wmma16(qa0, load_fragB(Kb + (size_t)(kt + 16) * HDIM + 0,  HDIM, lane), z));

    float p0[8], p1[8], corr[8];
#pragma unroll
    for (int jj = 0; jj < 8; ++jj) {
      float rm = redmax16(fmaxf(s0[jj], s1[jj]));
      float nm = fmaxf(mrow[jj], rm);
      corr[jj] = __expf(mrow[jj] - nm);
      mrow[jj] = nm;
      p0[jj] = __expf(s0[jj] - nm);
      p1[jj] = __expf(s1[jj] - nm);
      lrow[jj] = lrow[jj] * corr[jj] + redsum16(p0[jj] + p1[jj]);
    }
#pragma unroll
    for (int jj = 0; jj < 8; ++jj) {
      o0[jj] *= corr[jj]; o1[jj] *= corr[jj]; o2[jj] *= corr[jj]; o3[jj] *= corr[jj];
    }
    _Float16* pw = &Plds[wid][0];
#pragma unroll
    for (int jj = 0; jj < 8; ++jj) {
      const int row = jj + 8 * hi;
      pw[row * 32 + nl]      = (_Float16)p0[jj];
      pw[row * 32 + nl + 16] = (_Float16)p1[jj];
    }
    asm volatile("s_wait_dscnt 0" ::: "memory");
    v16h pa = load_fragA(pw, 32, lane);
    o0 = wmma16(pa, load_fragB(Vb + (size_t) 0 * SEQ + kt, SEQ, lane), o0);
    o1 = wmma16(pa, load_fragB(Vb + (size_t)16 * SEQ + kt, SEQ, lane), o1);
    o2 = wmma16(pa, load_fragB(Vb + (size_t)32 * SEQ + kt, SEQ, lane), o2);
    o3 = wmma16(pa, load_fragB(Vb + (size_t)48 * SEQ + kt, SEQ, lane), o3);
  }

  v8f oo[4] = {o0, o1, o2, o3};
#pragma unroll
  for (int j = 0; j < 4; ++j) {
    const int col = h * HDIM + j * 16 + nl;
#pragma unroll
    for (int jj = 0; jj < 8; ++jj) {
      const int gr = b * SEQ + mt * 16 + jj + 8 * hi;
      attnh[(size_t)gr * CH + col] = (_Float16)(oo[j][jj] / lrow[jj]);
    }
  }
}

// ---------------- projection GEMM: same TDM-staged panel scheme, fp32 output ----------------
__global__ void __launch_bounds__(256) k_proj_gemm(
    const _Float16* __restrict__ ah,   // [BN][CH]
    const _Float16* __restrict__ wT,   // [CH][CH]
    const float*    __restrict__ bias,
    float* __restrict__ out) {
  extern __shared__ _Float16 Bp[];
  const int NTN = CH / 64;                 // 16 column-panels
  const int tn  = blockIdx.x % NTN;
  const int tm  = (blockIdx.x / NTN) * 8 + (threadIdx.x >> 5);
  const int n0  = tn * 64;
  const int lane = threadIdx.x & 31;
  const int hi = lane >> 4, nl = lane & 15;

  stage_panel(wT + (size_t)n0 * CH, Bp, CH);

  v8f acc[4] = {};
  gemm_tile_lds(ah + (size_t)(tm * 16) * CH, Bp, lane, acc);

#pragma unroll
  for (int j = 0; j < 4; ++j) {
    const int c = n0 + j * 16 + nl;
    const float bv = bias[c];
#pragma unroll
    for (int jj = 0; jj < 8; ++jj) {
      const int row = tm * 16 + jj + 8 * hi;
      out[(size_t)row * CH + c] = acc[j][jj] + bv;
    }
  }
}

extern "C" void kernel_launch(void* const* d_in, const int* in_sizes, int n_in,
                              void* d_out, int out_size, void* d_ws, size_t ws_size,
                              hipStream_t stream) {
  const float* x      = (const float*)d_in[0];
  const float* w_qkv  = (const float*)d_in[1];
  const float* b_qkv  = (const float*)d_in[2];
  const float* w_proj = (const float*)d_in[3];
  const float* b_proj = (const float*)d_in[4];
  float* out = (float*)d_out;
  (void)in_sizes; (void)n_in; (void)out_size; (void)ws_size;

  char* ws = (char*)d_ws;
  size_t off = 0;
  auto alloc = [&](size_t bytes) -> char* {
    char* p = ws + off; off += (bytes + 255) & ~(size_t)255; return p;
  };
  _Float16* xh     = (_Float16*)alloc((size_t)BN * CH * 2);
  _Float16* wqkvT  = (_Float16*)alloc((size_t)QKV_N * CH * 2);
  _Float16* wprojT = (_Float16*)alloc((size_t)CH * CH * 2);
  _Float16* Qd     = (_Float16*)alloc((size_t)BN * CH * 2);
  _Float16* Kd     = (_Float16*)alloc((size_t)BN * CH * 2);
  _Float16* Vt     = (_Float16*)alloc((size_t)BN * CH * 2);
  _Float16* attnh  = (_Float16*)alloc((size_t)BN * CH * 2);

  const int n1 = BN * CH;
  k_cvt_f16<<<(n1 + 255) / 256, 256, 0, stream>>>(x, xh, n1);
  const int n2 = CH * QKV_N;
  k_transpose_f16<<<(n2 + 255) / 256, 256, 0, stream>>>(w_qkv, wqkvT, CH, QKV_N);
  const int n3 = CH * CH;
  k_transpose_f16<<<(n3 + 255) / 256, 256, 0, stream>>>(w_proj, wprojT, CH, CH);

  k_qkv_gemm<<<(BN / 16 / 8) * (QKV_N / 64), 256, PANEL_LDS_BYTES, stream>>>(
      xh, wqkvT, b_qkv, Qd, Kd, Vt);
  k_flash<<<(BATCH * HEADS * (SEQ / 16)) / 8, 256, 0, stream>>>(Qd, Kd, Vt, attnh);
  k_proj_gemm<<<(BN / 16 / 8) * (CH / 64), 256, PANEL_LDS_BYTES, stream>>>(
      attnh, wprojT, b_proj, out);
}
